// MHAtt_12756052869316
// MI455X (gfx1250) — matrix operand order repaired and checked by reading, not compiled, not run on hardware
//
#include <hip/hip_runtime.h>
#include <hip/hip_bf16.h>

// MHA forward on gfx1250 (MI455X): all matmuls via v_wmma_f32_16x16x32_f16.
// Compute-bound (~103 GFLOP vs ~150MB traffic): f16 WMMA with f32 accum.
// Data movement uses CDNA5 paths: global_load_async_to_lds_b128 (ASYNCcnt)
// for conversion-free tiles, TDM tensor_load_to_lds (TENSORcnt) for the
// attention K chunks, global_prefetch_b8 for streamed operands.

#define HSd 1024
#define Dd  64
#define Hn  16
#define Bn  8
#define Sn  1024
#define SCALE 0.125f  // 1/sqrt(64)

typedef _Float16 half_t;
typedef __attribute__((ext_vector_type(16))) _Float16 v16h;
typedef __attribute__((ext_vector_type(8)))  _Float16 v8h;
typedef __attribute__((ext_vector_type(4)))  _Float16 v4h;
typedef __attribute__((ext_vector_type(8)))  float    v8f;
typedef __attribute__((ext_vector_type(4)))  unsigned int v4u;
typedef __attribute__((ext_vector_type(8)))  int v8i;
typedef __attribute__((ext_vector_type(4)))  int v4i;

__device__ __forceinline__ v8f wmma_f16(v16h a, v16h b, v8f c) {
  return __builtin_amdgcn_wmma_f32_16x16x32_f16(false, a, false, b, (short)0, c,
                                                false, false);
}

__device__ __forceinline__ v16h pack16(v8h lo, v8h hi) {
  v16h r;
#pragma unroll
  for (int e = 0; e < 8; ++e) { r[e] = lo[e]; r[e + 8] = hi[e]; }
  return r;
}

// Async global->LDS 16B copy (no VGPR data path). LDS offset = low 32 bits of
// the flat shared-pointer (aperture keeps offset in addr[31:0]).
__device__ __forceinline__ void async_b128_to_lds(void* lds_dst, const void* gsrc) {
  unsigned loff = (unsigned)(uintptr_t)lds_dst;
  unsigned long long ga = (unsigned long long)(uintptr_t)gsrc;
  asm volatile("global_load_async_to_lds_b128 %0, %1, off"
               :: "v"(loff), "v"(ga) : "memory");
}
__device__ __forceinline__ void wait_async0() {
  asm volatile("s_wait_asynccnt 0x0" ::: "memory");
}

// ---------------------------------------------------------------- convert
__global__ __launch_bounds__(256) void f32_to_f16_kernel(
    const float* __restrict__ src, half_t* __restrict__ dst, int n) {
  int i = (blockIdx.x * 256 + threadIdx.x) * 4;
  if (i + 3 < n) {
    float4 v = *(const float4*)(src + i);
    v4h o;
    o[0] = (half_t)v.x; o[1] = (half_t)v.y; o[2] = (half_t)v.z; o[3] = (half_t)v.w;
    *(v4h*)(dst + i) = o;
  }
}

// ---------------------------------------------------------------- QKV proj
// Y = X @ W^T + bias; block tile 128(M) x 64(N), 8 waves x (16x64).
// X tile: f32 -> f16 through VGPRs (needs v_cvt). W tile: async -> LDS.
#define XSTR 40
#define WSTR 40
__global__ __launch_bounds__(256) void proj_qkv_kernel(
    const float* __restrict__ X, const half_t* __restrict__ W,
    const float* __restrict__ bias, half_t* __restrict__ Yh) {
  __shared__ half_t xs[128 * XSTR];
  __shared__ half_t wtile[64 * WSTR];
  const int t = threadIdx.x;
  const int lane = t & 31;
  const int w = t >> 5;
  const int l15 = lane & 15;
  const int hi16 = lane >> 4;
  const int mBase = blockIdx.x * 128;
  const int nBase = blockIdx.y * 64;

  v8f acc[4];
#pragma unroll
  for (int nt = 0; nt < 4; ++nt)
#pragma unroll
    for (int r = 0; r < 8; ++r) acc[nt][r] = 0.f;

  for (int kb = 0; kb < HSd; kb += 32) {
    __syncthreads();
    {  // W 64x32 f16: async global->LDS, one b128 per thread
      const int i = t >> 2, j8 = (t & 3) * 8;
      async_b128_to_lds(&wtile[i * WSTR + j8],
                        W + (size_t)(nBase + i) * HSd + kb + j8);
    }
    {  // X 128x32: f32 load + convert + ds_store
      const int i0 = t >> 3, j4 = (t & 7) * 4;
#pragma unroll
      for (int it = 0; it < 4; ++it) {
        int i = i0 + it * 32;
        float4 xv = *(const float4*)(X + (size_t)(mBase + i) * HSd + kb + j4);
        half_t* p = &xs[i * XSTR + j4];
        p[0] = (half_t)xv.x; p[1] = (half_t)xv.y;
        p[2] = (half_t)xv.z; p[3] = (half_t)xv.w;
      }
    }
    if (kb + 32 < HSd)
      __builtin_prefetch(X + (size_t)(mBase + (t >> 3)) * HSd + kb + 32, 0, 1);
    wait_async0();
    __syncthreads();
    // A fragment: lane = M row; K split per ISA 16-bit A layout
    const half_t* arow = &xs[(w * 16 + l15) * XSTR];
    const int ab = hi16 * 8;
    v16h a = pack16(*(const v8h*)(arow + ab), *(const v8h*)(arow + ab + 16));
#pragma unroll
    for (int nt = 0; nt < 4; ++nt) {
      const half_t* brow = &wtile[(nt * 16 + l15) * WSTR + hi16 * 8];
      v16h bf = pack16(*(const v8h*)brow, *(const v8h*)(brow + 16));
      acc[nt] = wmma_f16(a, bf, acc[nt]);
    }
  }
#pragma unroll
  for (int nt = 0; nt < 4; ++nt)
#pragma unroll
    for (int r = 0; r < 8; ++r) {
      int m = mBase + w * 16 + r + 8 * hi16;
      int n = nBase + nt * 16 + l15;
      float val = acc[nt][r] + bias[n];
      int b = m >> 10, s = m & (Sn - 1);
      int h = n >> 6, d = n & (Dd - 1);
      Yh[(((size_t)b * Hn + h) * Sn + s) * Dd + d] = (half_t)val;
    }
}

// ---------------------------------------------------------------- out proj
// ctx (f16) @ Wo^T + bo -> f32. Both tiles are pure copies -> all async.
__global__ __launch_bounds__(256) void out_proj_kernel(
    const half_t* __restrict__ X, const half_t* __restrict__ W,
    const float* __restrict__ bias, float* __restrict__ out) {
  __shared__ half_t xs[128 * XSTR];
  __shared__ half_t wtile[64 * WSTR];
  const int t = threadIdx.x;
  const int lane = t & 31;
  const int w = t >> 5;
  const int l15 = lane & 15;
  const int hi16 = lane >> 4;
  const int mBase = blockIdx.x * 128;
  const int nBase = blockIdx.y * 64;

  v8f acc[4];
#pragma unroll
  for (int nt = 0; nt < 4; ++nt)
#pragma unroll
    for (int r = 0; r < 8; ++r) acc[nt][r] = 0.f;

  for (int kb = 0; kb < HSd; kb += 32) {
    __syncthreads();
    {
      const int i0 = t >> 2, j8 = (t & 3) * 8;
#pragma unroll
      for (int it = 0; it < 2; ++it) {
        int i = i0 + it * 64;
        async_b128_to_lds(&xs[i * XSTR + j8],
                          X + (size_t)(mBase + i) * HSd + kb + j8);
      }
      async_b128_to_lds(&wtile[i0 * WSTR + j8],
                        W + (size_t)(nBase + i0) * HSd + kb + j8);
    }
    wait_async0();
    __syncthreads();
    const half_t* arow = &xs[(w * 16 + l15) * XSTR];
    const int ab = hi16 * 8;
    v16h a = pack16(*(const v8h*)(arow + ab), *(const v8h*)(arow + ab + 16));
#pragma unroll
    for (int nt = 0; nt < 4; ++nt) {
      const half_t* brow = &wtile[(nt * 16 + l15) * WSTR + hi16 * 8];
      v16h bf = pack16(*(const v8h*)brow, *(const v8h*)(brow + 16));
      acc[nt] = wmma_f16(a, bf, acc[nt]);
    }
  }
#pragma unroll
  for (int nt = 0; nt < 4; ++nt)
#pragma unroll
    for (int r = 0; r < 8; ++r) {
      int m = mBase + w * 16 + r + 8 * hi16;
      int n = nBase + nt * 16 + l15;
      out[(size_t)m * HSd + n] = acc[nt][r] + bias[n];
    }
}

// ---------------------------------------------------------------- attention
// One block = (b, h, 64 q-rows); 4 waves x 16 q-rows; flash online softmax.
// K chunk staged by the Tensor Data Mover (one 32x64 2D tile per chunk, with
// LDS padding 64->72 elements/row done by TDM pad fields); Q tile staged via
// async global->LDS; V chunk needs a transpose so it rides the VGPR path.
__global__ __launch_bounds__(128) void attention_kernel(
    const half_t* __restrict__ Qh, const half_t* __restrict__ Kh,
    const half_t* __restrict__ Vh, const unsigned char* __restrict__ mask,
    half_t* __restrict__ ctx) {
  __shared__ half_t q_lds[64 * 72];      // [qrow][d]
  __shared__ half_t k_lds[32 * 72];      // [key][d]   (TDM-padded rows)
  __shared__ half_t vt_lds[64 * 40];     // [d][key]
  __shared__ half_t p_lds[4 * 16 * 40];  // per-wave P repack

  const int t = threadIdx.x;
  const int lane = t & 31;
  const int w = t >> 5;
  const int l15 = lane & 15;
  const int hi16 = lane >> 4;

  const int blk = blockIdx.x;
  const int qt = blk & 15;
  const int h = (blk >> 4) & 15;
  const int b = blk >> 8;
  const int qBase = qt * 64;

  const size_t headOff = (((size_t)b * Hn + h) * Sn) * Dd;
  const half_t* Qp = Qh + headOff + (size_t)qBase * Dd;
  const half_t* Kp = Kh + headOff;
  const half_t* Vp = Vh + headOff;
  const unsigned char* mp = mask + (size_t)b * Sn;

  {  // stage Q tile 64x64 once via async copies
    const int i0 = t >> 3, j8 = (t & 7) * 8;
#pragma unroll
    for (int it = 0; it < 4; ++it) {
      int i = i0 + it * 16;
      async_b128_to_lds(&q_lds[i * 72 + j8], Qp + (size_t)i * Dd + j8);
    }
    wait_async0();
  }

  float mrow[8], lrow[8];
  v8f o[4];
#pragma unroll
  for (int r = 0; r < 8; ++r) { mrow[r] = -1e30f; lrow[r] = 0.f; }
#pragma unroll
  for (int nd = 0; nd < 4; ++nd)
#pragma unroll
    for (int r = 0; r < 8; ++r) o[nd][r] = 0.f;

  for (int kc = 0; kc < Sn; kc += 32) {
    __syncthreads();
#if __has_builtin(__builtin_amdgcn_tensor_load_to_lds)
    if (t < 32) {  // wave 0 issues one TDM op for the whole K chunk
      unsigned lds0 = (unsigned)(uintptr_t)&k_lds[0];
      unsigned long long ga =
          (unsigned long long)(uintptr_t)(Kp + (size_t)kc * Dd);
      v4u g0;
      g0[0] = 1u;                                   // count=1 (valid user D#)
      g0[1] = lds0;                                 // lds_addr
      g0[2] = (unsigned)(ga & 0xffffffffu);         // global_addr[31:0]
      g0[3] = (unsigned)((ga >> 32) & 0x01ffffffu)  // global_addr[56:32]
              | (2u << 30);                         // type=2 (image)
      v8i g1;
      g1[0] = (int)((1u << 16)     // data_size = 2B
                    | (1u << 20)   // pad_enable
                    | (4u << 22)   // pad_interval: 32 DWORDs (=64 f16 = 1 row)
                    | (3u << 25)); // pad_amount: 4 DWORDs (=8 f16) -> stride 72
      g1[1] = (int)(64u << 16);    // tensor_dim0 = 64 (low half in [31:16])
      g1[2] = (int)(1024u << 16);  // tensor_dim1 = 1024 (low half in [31:16])
      g1[3] = (int)(64u << 16);    // tile_dim0 = 64
      g1[4] = 32;                  // tile_dim1 = 32 keys
      g1[5] = 64;                  // tensor_dim0_stride = 64 elements
      g1[6] = 0; g1[7] = 0;
      v4i g2; g2[0] = 0; g2[1] = 0; g2[2] = 0; g2[3] = 0;
      v4i g3 = g2;
      v8i g4;  // extra descriptor group in the 6-arg toolchain variant; unused
      g4[0] = 0; g4[1] = 0; g4[2] = 0; g4[3] = 0;
      g4[4] = 0; g4[5] = 0; g4[6] = 0; g4[7] = 0;
      __builtin_amdgcn_tensor_load_to_lds(g0, g1, g2, g3, g4, 0);
      __builtin_amdgcn_s_wait_tensorcnt(0);
    }
#else
    {  // fallback: async copies for K chunk
      const int i0 = t >> 3, j8 = (t & 7) * 8;
#pragma unroll
      for (int it = 0; it < 2; ++it) {
        int key = i0 + it * 16;
        async_b128_to_lds(&k_lds[key * 72 + j8],
                          Kp + (size_t)(kc + key) * Dd + j8);
      }
      wait_async0();
    }
#endif
    {  // V chunk transposed into [64][32] via VGPRs
      const int i0 = t >> 3, j8 = (t & 7) * 8;
#pragma unroll
      for (int it = 0; it < 2; ++it) {
        int key = i0 + it * 16;
        v8h vv = *(const v8h*)(Vp + (size_t)(kc + key) * Dd + j8);
#pragma unroll
        for (int e = 0; e < 8; ++e) vt_lds[(j8 + e) * 40 + key] = vv[e];
      }
    }
    if (kc + 32 < Sn)
      __builtin_prefetch(Vp + (size_t)(kc + 32 + (t >> 3)) * Dd, 0, 1);
    __syncthreads();

    // scores: two 16x16 tiles over keys [kc, kc+32)
    v8f s0, s1;
#pragma unroll
    for (int r = 0; r < 8; ++r) { s0[r] = 0.f; s1[r] = 0.f; }
#pragma unroll
    for (int dh = 0; dh < 2; ++dh) {
      const half_t* arow = &q_lds[(w * 16 + l15) * 72 + dh * 32];
      const int ab = hi16 * 8;
      v16h a = pack16(*(const v8h*)(arow + ab), *(const v8h*)(arow + ab + 16));
#pragma unroll
      for (int ktile = 0; ktile < 2; ++ktile) {
        const half_t* brow = &k_lds[(ktile * 16 + l15) * 72 + dh * 32 + hi16 * 8];
        v16h bf = pack16(*(const v8h*)brow, *(const v8h*)(brow + 16));
        if (ktile == 0) s0 = wmma_f16(a, bf, s0);
        else            s1 = wmma_f16(a, bf, s1);
      }
    }

    // scale + mask (mask==true -> -1e9) + online softmax
    const bool msk0 = mp[kc + l15] != 0;
    const bool msk1 = mp[kc + l15 + 16] != 0;
    float al[8];
#pragma unroll
    for (int r = 0; r < 8; ++r) {
      float v0 = msk0 ? -1e9f : s0[r] * SCALE;
      float v1 = msk1 ? -1e9f : s1[r] * SCALE;
      float cm = fmaxf(v0, v1);
#pragma unroll
      for (int off = 8; off >= 1; off >>= 1)
        cm = fmaxf(cm, __shfl_xor(cm, off, 32));
      float nm = fmaxf(mrow[r], cm);
      float a0 = __expf(mrow[r] - nm);
      float p0 = __expf(v0 - nm);
      float p1 = __expf(v1 - nm);
      float rs = p0 + p1;
#pragma unroll
      for (int off = 8; off >= 1; off >>= 1)
        rs += __shfl_xor(rs, off, 32);
      lrow[r] = lrow[r] * a0 + rs;
      mrow[r] = nm;
      al[r] = a0;
      s0[r] = p0;
      s1[r] = p1;
    }
#pragma unroll
    for (int nd = 0; nd < 4; ++nd)
#pragma unroll
      for (int r = 0; r < 8; ++r) o[nd][r] *= al[r];

    // P: C layout -> A layout via per-wave LDS bounce
    half_t* pw = &p_lds[w * 16 * 40];
#pragma unroll
    for (int r = 0; r < 8; ++r) {
      pw[(r + 8 * hi16) * 40 + l15] = (half_t)s0[r];
      pw[(r + 8 * hi16) * 40 + l15 + 16] = (half_t)s1[r];
    }
    const half_t* prow = &p_lds[(w * 16 + l15) * 40];
    const int ab = hi16 * 8;
    v16h pa = pack16(*(const v8h*)(prow + ab), *(const v8h*)(prow + ab + 16));
#pragma unroll
    for (int nd = 0; nd < 4; ++nd) {
      const half_t* vrow = &vt_lds[(nd * 16 + l15) * 40 + hi16 * 8];
      v16h vb = pack16(*(const v8h*)vrow, *(const v8h*)(vrow + 16));
      o[nd] = wmma_f16(pa, vb, o[nd]);
    }
  }

  // normalize and write ctx [B,S,HS] f16
#pragma unroll
  for (int r = 0; r < 8; ++r) {
    float inv = 1.0f / lrow[r];
    int srow = qBase + w * 16 + r + 8 * hi16;
#pragma unroll
    for (int nd = 0; nd < 4; ++nd) {
      int d = nd * 16 + l15;
      ctx[((size_t)b * Sn + srow) * HSd + h * Dd + d] = (half_t)(o[nd][r] * inv);
    }
  }
}

// ---------------------------------------------------------------- launch
extern "C" void kernel_launch(void* const* d_in, const int* in_sizes, int n_in,
                              void* d_out, int out_size, void* d_ws, size_t ws_size,
                              hipStream_t stream) {
  const float* v  = (const float*)d_in[0];
  const float* k  = (const float*)d_in[1];
  const float* q  = (const float*)d_in[2];
  const unsigned char* mask = (const unsigned char*)d_in[3];
  const float* Wq = (const float*)d_in[4];
  const float* bq = (const float*)d_in[5];
  const float* Wk = (const float*)d_in[6];
  const float* bk = (const float*)d_in[7];
  const float* Wv = (const float*)d_in[8];
  const float* bv = (const float*)d_in[9];
  const float* Wo = (const float*)d_in[10];
  const float* bo = (const float*)d_in[11];
  float* out = (float*)d_out;

  char* ws = (char*)d_ws;
  half_t* Wq16 = (half_t*)(ws + ((size_t)0 << 20));
  half_t* Wk16 = (half_t*)(ws + ((size_t)2 << 20));
  half_t* Wv16 = (half_t*)(ws + ((size_t)4 << 20));
  half_t* Wo16 = (half_t*)(ws + ((size_t)6 << 20));
  half_t* Qh   = (half_t*)(ws + ((size_t)8 << 20));
  half_t* Kh   = (half_t*)(ws + ((size_t)24 << 20));
  half_t* Vh   = (half_t*)(ws + ((size_t)40 << 20));
  half_t* ctx  = (half_t*)(ws + ((size_t)56 << 20));  // total 72 MB

  const int NW = HSd * HSd;
  dim3 cgrid(NW / (256 * 4)), cblk(256);
  f32_to_f16_kernel<<<cgrid, cblk, 0, stream>>>(Wq, Wq16, NW);
  f32_to_f16_kernel<<<cgrid, cblk, 0, stream>>>(Wk, Wk16, NW);
  f32_to_f16_kernel<<<cgrid, cblk, 0, stream>>>(Wv, Wv16, NW);
  f32_to_f16_kernel<<<cgrid, cblk, 0, stream>>>(Wo, Wo16, NW);

  dim3 pgrid(Bn * Sn / 128, HSd / 64);  // 64 x 16 blocks
  proj_qkv_kernel<<<pgrid, 256, 0, stream>>>(q, Wq16, bq, Qh);
  proj_qkv_kernel<<<pgrid, 256, 0, stream>>>(k, Wk16, bk, Kh);
  proj_qkv_kernel<<<pgrid, 256, 0, stream>>>(v, Wv16, bv, Vh);

  attention_kernel<<<dim3(Bn * Hn * (Sn / 64)), 128, 0, stream>>>(Qh, Kh, Vh, mask, ctx);

  out_proj_kernel<<<pgrid, 256, 0, stream>>>(ctx, Wo16, bo, out);
}
